// RWKVRNNCell_2654289789469
// MI455X (gfx1250) — matrix operand order repaired and checked by reading, not compiled, not run on hardware
//
#include <hip/hip_runtime.h>
#include <math.h>

// ---------------------------------------------------------------------------
// RWKV block (B=8, T=512, D=H=768) for MI455X / gfx1250.
// Persistent cooperative kernel: 48 WGs x 128 threads, 3 grid barriers/step.
// GEMMs via V_WMMA_F32_16X16X32_BF16, weights pre-packed to bf16 fragments.
// A-fragments: two contiguous 16B chunks per lane (ISA 7.12.2 layout); pad
// rows (M=8..15) read duplicated rows 0..7 -> garbage C rows that are unread.
// ---------------------------------------------------------------------------

typedef __attribute__((ext_vector_type(16))) __bf16 v16bf;
typedef __attribute__((ext_vector_type(8)))  __bf16 v8bf;
typedef __attribute__((ext_vector_type(4)))  __bf16 v4bf;
typedef __attribute__((ext_vector_type(8)))  float  v8f;

#define Bsz 8
#define Tsz 512
#define Dsz 768
#define Hsz 768
#define H4  3072
#define NWG 48
#define NTHREADS 128
#define NKB_D  24   // 768/32
#define NKB_4H 96   // 3072/32
#define LN_EPS 1e-3f

__device__ inline float fast_rcp(float x) { return __builtin_amdgcn_rcpf(x); }

// ---- load one lane's A fragment from a natural row-major bf16 row ---------
// lane hi=0: K = K0+0..7 and K0+16..23 ; lane hi=1: +8 (K0 includes hi*8)
__device__ inline v16bf ld_nat(const __bf16* __restrict__ rowp, int K0) {
  v8bf lo = *(const v8bf*)(rowp + K0);
  v8bf hi = *(const v8bf*)(rowp + K0 + 16);
  return __builtin_shufflevector(lo, hi, 0, 1, 2, 3, 4, 5, 6, 7,
                                 8, 9, 10, 11, 12, 13, 14, 15);
}

// ---- GEMM tile, A read directly from a bf16 buffer (rows 0..7 valid) ------
__device__ inline v8f gemm_direct(const __bf16* __restrict__ Wp, int nkb, int tile,
                                  int kb0, int kbStep, int lane,
                                  const __bf16* __restrict__ A, int ldK, v8f c) {
  const int rowc = lane & 7;            // pad rows duplicate rows 0..7
  const int hi   = (lane >> 4) & 1;
  const __bf16* rowp = A + rowc * ldK;
  for (int kb = kb0; kb < nkb; kb += kbStep) {
    const __bf16* bp = Wp + (((size_t)tile * nkb + kb) * 32 + lane) * 16;
    __builtin_prefetch(bp + (size_t)kbStep * 512, 0, 1);
    v16bf b = *(const v16bf*)bp;
    v16bf a = ld_nat(rowp, kb * 32 + hi * 8);
    c = __builtin_amdgcn_wmma_f32_16x16x32_bf16(false, a, false, b, (short)0, c,
                                                false, false);
  }
  return c;
}

// ---- GEMM tile with on-the-fly token mixing: a = xo + m*(xc - xo) ---------
__device__ inline v8f gemm_mixed(const __bf16* __restrict__ Wp, int nkb, int tile,
                                 int kb0, int kbStep, int lane,
                                 const __bf16* __restrict__ Xc,
                                 const __bf16* __restrict__ Xo,
                                 const __bf16* __restrict__ Mx, v8f c) {
  const int rowc = lane & 7;
  const int hi   = (lane >> 4) & 1;
  const __bf16* cp = Xc + rowc * Dsz;
  const __bf16* op = Xo + rowc * Dsz;
  for (int kb = kb0; kb < nkb; kb += kbStep) {
    const __bf16* bp = Wp + (((size_t)tile * nkb + kb) * 32 + lane) * 16;
    __builtin_prefetch(bp + (size_t)kbStep * 512, 0, 1);
    v16bf b = *(const v16bf*)bp;
    const int K0 = kb * 32 + hi * 8;
    v16bf xc_ = ld_nat(cp, K0);
    v16bf xo_ = ld_nat(op, K0);
    v16bf m_  = ld_nat(Mx, K0);
    v16bf a;
#pragma unroll
    for (int e = 0; e < 16; ++e) {
      float xo = (float)xo_[e];
      a[e] = (__bf16)(xo + (float)m_[e] * ((float)xc_[e] - xo));
    }
    c = __builtin_amdgcn_wmma_f32_16x16x32_bf16(false, a, false, b, (short)0, c,
                                                false, false);
  }
  return c;
}

// ---- weight fp32 -> bf16, packed in B-fragment order ----------------------
// B 32x16 layout: col=lane&15, hi=lane>>4; e<8 -> K=kb*32+hi*8+e ; e>=8 -> K=kb*32+16+hi*8+(e-8)
__global__ void pack_weight_bf16(const float* __restrict__ src,
                                 __bf16* __restrict__ dst, int K, int N) {
  int idx = blockIdx.x * blockDim.x + threadIdx.x;
  if (idx >= K * N) return;
  int e    = idx & 15;
  int lane = (idx >> 4) & 31;
  int blk  = idx >> 9;            // tile*nkb + kb
  int nkb  = K >> 5;
  int kb   = blk % nkb;
  int tile = blk / nkb;
  int col  = tile * 16 + (lane & 15);
  int k    = kb * 32 + ((e >> 3) * 16) + ((lane >> 4) * 8) + (e & 7);
  dst[idx] = (__bf16)src[(size_t)k * N + col];
}

// ---- device-wide sense barrier (bar[0]=count, bar[1]=generation) ----------
__device__ inline void grid_sync(int* bar, int nwg) {
  __threadfence();
  __syncthreads();
  if (threadIdx.x == 0) {
    volatile int* vbar = bar;
    int g = vbar[1];
    if (atomicAdd(&bar[0], 1) == nwg - 1) {
      bar[0] = 0;
      __threadfence();
      atomicAdd(&bar[1], 1);
    } else {
      while (vbar[1] == g) __builtin_amdgcn_s_sleep(1);
    }
  }
  __syncthreads();
  __threadfence();
}

// ---- layernorm of 8 rows (row = tid>>4, 16 threads/row) -> bf16 LDS -------
// base/rowStride give f32 source rows; float4 vector scan for coalescing.
__device__ inline void layernorm_to_lds(const float* __restrict__ base,
                                        size_t rowStride,
                                        const float* __restrict__ g,
                                        const float* __restrict__ bb,
                                        __bf16* dst, int tid) {
  const int row = tid >> 4, j16 = tid & 15;
  const float4* rp = (const float4*)(base + (size_t)row * rowStride);
  const float4* g4 = (const float4*)g;
  const float4* b4 = (const float4*)bb;
  float s = 0.f, s2 = 0.f;
  for (int j = j16; j < Dsz / 4; j += 16) {
    float4 v = rp[j];
    s  += v.x + v.y + v.z + v.w;
    s2 += v.x * v.x + v.y * v.y + v.z * v.z + v.w * v.w;
  }
#pragma unroll
  for (int m = 8; m >= 1; m >>= 1) {
    s  += __shfl_xor(s,  m, 16);
    s2 += __shfl_xor(s2, m, 16);
  }
  float mean = s * (1.f / Dsz);
  float var  = s2 * (1.f / Dsz) - mean * mean;
  float rstd = rsqrtf(var + LN_EPS);
  for (int j = j16; j < Dsz / 4; j += 16) {
    float4 v = rp[j];
    float4 gg = g4[j], bbv = b4[j];
    v4bf o;
    o[0] = (__bf16)((v.x - mean) * rstd * gg.x + bbv.x);
    o[1] = (__bf16)((v.y - mean) * rstd * gg.y + bbv.y);
    o[2] = (__bf16)((v.z - mean) * rstd * gg.z + bbv.z);
    o[3] = (__bf16)((v.w - mean) * rstd * gg.w + bbv.w);
    *(v4bf*)(dst + row * Dsz + j * 4) = o;
  }
}

__global__ __launch_bounds__(NTHREADS)
void rwkv_persistent(const float* __restrict__ x,
                     const float* __restrict__ ln1_g, const float* __restrict__ ln1_b,
                     const float* __restrict__ ln2_g, const float* __restrict__ ln2_b,
                     const float* __restrict__ tm_mix_k, const float* __restrict__ tm_mix_v,
                     const float* __restrict__ tm_mix_r,
                     const float* __restrict__ time_decay, const float* __restrict__ time_first,
                     const float* __restrict__ cm_mix_k, const float* __restrict__ cm_mix_r,
                     const __bf16* __restrict__ Wtmk, const __bf16* __restrict__ Wtmv,
                     const __bf16* __restrict__ Wtmr, const __bf16* __restrict__ Wout,
                     const __bf16* __restrict__ Wcmk, const __bf16* __restrict__ Wcmr,
                     const __bf16* __restrict__ Wcmv,
                     __bf16* __restrict__ srwkvB, float* __restrict__ ybufB,
                     __bf16* __restrict__ r2B,
                     int* bar, float* __restrict__ out) {
  __shared__ __bf16 xiBuf[2][Bsz * Dsz];   // xi ping-pong (cur / prev = s_tm)
  __shared__ __bf16 xcBuf[2][Bsz * Dsz];   // xc ping-pong (cur / prev = s_cm)
  __shared__ __bf16 mixS[5][Dsz];          // tmk,tmv,tmr,cmk,cmr
  __shared__ float  scratch[512];          // kvr (384f) / wave partials (512f)
  __shared__ float  ylocal[Bsz * 16];
  __shared__ float  sigr[Bsz * 16];

  const int tid  = threadIdx.x;
  const int wg   = blockIdx.x;
  const int lane = tid & 31;
  const int wave = tid >> 5;
  const int b_   = tid >> 4;      // 0..7
  const int hl   = tid & 15;      // 0..15
  const int col  = wg * 16 + hl;  // this WG's H/D column slice

  for (int i = tid; i < Dsz; i += NTHREADS) {
    mixS[0][i] = (__bf16)tm_mix_k[i];
    mixS[1][i] = (__bf16)tm_mix_v[i];
    mixS[2][i] = (__bf16)tm_mix_r[i];
    mixS[3][i] = (__bf16)cm_mix_k[i];
    mixS[4][i] = (__bf16)cm_mix_r[i];
  }
  for (int i = tid; i < Bsz * Dsz; i += NTHREADS) {
    xiBuf[1][i] = (__bf16)0.f;   // s_tm(t=0) = 0
    xcBuf[1][i] = (__bf16)0.f;   // s_cm(t=0) = 0
  }
  // per-(b,h) recurrent state + hoisted per-column constants
  float q = 0.f, num = 0.f, den = 0.f;
  const float tf = time_first[col];
  const float ed = __expf(time_decay[col]);
  __syncthreads();

  for (int t = 0; t < Tsz; ++t) {
    const int pp = t & 1;
    __bf16* xiC = xiBuf[pp];     __bf16* xiO = xiBuf[pp ^ 1];
    __bf16* xcC = xcBuf[pp];     __bf16* xcO = xcBuf[pp ^ 1];

    // ---- Phase A: xi = LN1(x_t), redundant per WG -------------------------
    layernorm_to_lds(x + (size_t)t * Dsz, (size_t)Tsz * Dsz,
                     ln1_g, ln1_b, xiC, tid);
    __syncthreads();

    // ---- Phase B: k / v / r tiles (waves 0..2, full K) --------------------
    if (wave < 3) {
      const __bf16* W = (wave == 0) ? Wtmk : (wave == 1) ? Wtmv : Wtmr;
      v8f c = {};
      c = gemm_mixed(W, NKB_D, wg, 0, 1, lane, xiC, xiO, mixS[wave], c);
      if (lane < 16) {
#pragma unroll
        for (int i = 0; i < 8; ++i) scratch[(wave * 8 + i) * 16 + lane] = c[i];
      }
    }
    __syncthreads();

    // ---- Phase C: WKV recurrence (state in registers) ---------------------
    {
      float k = scratch[(0 * 8 + b_) * 16 + hl];
      float v = scratch[(1 * 8 + b_) * 16 + hl];
      float r = scratch[(2 * 8 + b_) * 16 + hl];
      float sr = fast_rcp(1.f + __expf(-r));
      float w  = tf + k;
      float qq = fmaxf(q, w);
      float e1 = __expf(q - qq), e2 = __expf(w - qq);
      float wkv = (e1 * num + e2 * v) * fast_rcp(e1 * den + e2);
      float w2 = q - ed;
      float qn = fmaxf(w2, k);
      e1 = __expf(w2 - qn); e2 = __expf(k - qn);
      num = e1 * num + e2 * v;
      den = e1 * den + e2;
      q = qn;
      srwkvB[b_ * Hsz + col] = (__bf16)(sr * wkv);
    }
    grid_sync(bar, NWG);   // barrier 1: srwkv row complete

    // ---- Phase D: y = x + (sr*wkv) @ out_w  (K split over 4 waves) --------
    {
      v8f c = {};
      c = gemm_direct(Wout, NKB_D, wg, wave, 4, lane, srwkvB, Hsz, c);
      if (lane < 16) {
#pragma unroll
        for (int i = 0; i < 8; ++i) scratch[(wave * 16 + lane) * 8 + i] = c[i];
      }
    }
    __syncthreads();
    {
      float s = 0.f;
#pragma unroll
      for (int w = 0; w < 4; ++w) s += scratch[(w * 16 + hl) * 8 + b_];
      float y = x[((size_t)b_ * Tsz + t) * Dsz + col] + s;
      ylocal[b_ * 16 + hl] = y;
      ybufB[b_ * Dsz + col] = y;
    }
    grid_sync(bar, NWG);   // barrier 2: y row complete

    // ---- Phase E: xc = LN2(y), redundant per WG ---------------------------
    layernorm_to_lds(ybufB, (size_t)Dsz, ln2_g, ln2_b, xcC, tid);
    __syncthreads();

    // ---- Phase F: relu(xk2@cm_key)^2 -> r2 ; sigmoid(xr2@cm_rec) ----------
    {
      int tile = wg * 4 + wave;          // 192 tiles of N=3072
      v8f c = {};
      c = gemm_mixed(Wcmk, NKB_D, tile, 0, 1, lane, xcC, xcO, mixS[3], c);
      if (lane < 16) {
        int cl = tile * 16 + lane;
#pragma unroll
        for (int i = 0; i < 8; ++i) {
          float rl = fmaxf(c[i], 0.f);
          r2B[i * H4 + cl] = (__bf16)(rl * rl);
        }
      }
    }
    {
      v8f c = {};
      c = gemm_mixed(Wcmr, NKB_D, wg, wave, 4, lane, xcC, xcO, mixS[4], c);
      __syncthreads();
      if (lane < 16) {
#pragma unroll
        for (int i = 0; i < 8; ++i) scratch[(wave * 16 + lane) * 8 + i] = c[i];
      }
    }
    __syncthreads();
    {
      float s = 0.f;
#pragma unroll
      for (int w = 0; w < 4; ++w) s += scratch[(w * 16 + hl) * 8 + b_];
      sigr[b_ * 16 + hl] = fast_rcp(1.f + __expf(-s));
    }
    grid_sync(bar, NWG);   // barrier 3: r2 row complete

    // ---- Phase G: kv = r2 @ cm_value ; out = y + sig*kv -------------------
    {
      v8f c = {};
      c = gemm_direct(Wcmv, NKB_4H, wg, wave, 4, lane, r2B, H4, c);
      __syncthreads();
      if (lane < 16) {
#pragma unroll
        for (int i = 0; i < 8; ++i) scratch[(wave * 16 + lane) * 8 + i] = c[i];
      }
    }
    __syncthreads();
    {
      float s = 0.f;
#pragma unroll
      for (int w = 0; w < 4; ++w) s += scratch[(w * 16 + hl) * 8 + b_];
      out[((size_t)b_ * Tsz + t) * Dsz + col] =
          ylocal[b_ * 16 + hl] + sigr[b_ * 16 + hl] * s;
    }
    __syncthreads();   // scratch reused by next step's phase B
  }
}

// ---------------------------------------------------------------------------
extern "C" void kernel_launch(void* const* d_in, const int* in_sizes, int n_in,
                              void* d_out, int out_size, void* d_ws, size_t ws_size,
                              hipStream_t stream) {
  (void)in_sizes; (void)n_in; (void)out_size; (void)ws_size;
  const float* x          = (const float*)d_in[0];
  const float* ln1_g      = (const float*)d_in[1];
  const float* ln1_b      = (const float*)d_in[2];
  const float* ln2_g      = (const float*)d_in[3];
  const float* ln2_b      = (const float*)d_in[4];
  const float* tm_mix_k   = (const float*)d_in[5];
  const float* tm_mix_v   = (const float*)d_in[6];
  const float* tm_mix_r   = (const float*)d_in[7];
  const float* tm_key_w   = (const float*)d_in[8];
  const float* tm_value_w = (const float*)d_in[9];
  const float* tm_rec_w   = (const float*)d_in[10];
  const float* time_decay = (const float*)d_in[11];
  const float* time_first = (const float*)d_in[12];
  const float* out_w      = (const float*)d_in[13];
  const float* cm_mix_k   = (const float*)d_in[14];
  const float* cm_mix_r   = (const float*)d_in[15];
  const float* cm_key_w   = (const float*)d_in[16];
  const float* cm_value_w = (const float*)d_in[17];
  const float* cm_rec_w   = (const float*)d_in[18];

  char* ws = (char*)d_ws;
  int* bar = (int*)ws;                       // 256 B barrier region
  size_t off = 256;
  const size_t EDxD  = (size_t)Dsz * Dsz * sizeof(__bf16);   // 1,179,648 B
  const size_t EDx4H = (size_t)Dsz * H4  * sizeof(__bf16);   // 4,718,592 B
  __bf16* Wtmk = (__bf16*)(ws + off); off += EDxD;
  __bf16* Wtmv = (__bf16*)(ws + off); off += EDxD;
  __bf16* Wtmr = (__bf16*)(ws + off); off += EDxD;
  __bf16* Wout = (__bf16*)(ws + off); off += EDxD;
  __bf16* Wcmk = (__bf16*)(ws + off); off += EDx4H;
  __bf16* Wcmr = (__bf16*)(ws + off); off += EDxD;
  __bf16* Wcmv = (__bf16*)(ws + off); off += EDx4H;
  __bf16* srwkvB = (__bf16*)(ws + off); off += (size_t)Bsz * Hsz * sizeof(__bf16);
  float*  ybufB  = (float*)(ws + off);  off += (size_t)Bsz * Dsz * sizeof(float);
  __bf16* r2B    = (__bf16*)(ws + off); off += (size_t)Bsz * H4 * sizeof(__bf16);

  hipMemsetAsync(bar, 0, 256, stream);

  const int thr = 256;
  auto pack = [&](const float* s, __bf16* d, int K, int N) {
    int total = K * N;
    pack_weight_bf16<<<(total + thr - 1) / thr, thr, 0, stream>>>(s, d, K, N);
  };
  pack(tm_key_w,   Wtmk, Dsz, Hsz);
  pack(tm_value_w, Wtmv, Dsz, Hsz);
  pack(tm_rec_w,   Wtmr, Dsz, Hsz);
  pack(out_w,      Wout, Hsz, Dsz);
  pack(cm_key_w,   Wcmk, Dsz, H4);
  pack(cm_rec_w,   Wcmr, Dsz, Dsz);
  pack(cm_value_w, Wcmv, H4,  Dsz);

  rwkv_persistent<<<NWG, NTHREADS, 0, stream>>>(
      x, ln1_g, ln1_b, ln2_g, ln2_b,
      tm_mix_k, tm_mix_v, tm_mix_r, time_decay, time_first,
      cm_mix_k, cm_mix_r,
      Wtmk, Wtmv, Wtmr, Wout, Wcmk, Wcmr, Wcmv,
      srwkvB, ybufB, r2B, bar, (float*)d_out);
}
